// ContextAwareTracker_57793079935101
// MI455X (gfx1250) — matrix operand order repaired and checked by reading, not compiled, not run on hardware
//
#include <hip/hip_runtime.h>
#include <hip/hip_bf16.h>

// ---------------------------------------------------------------------------
// LSTM encoder-decoder for gfx1250 (MI455X), bf16 WMMA mixed precision.
// B=16384, T=50, I=5, H=128, PRED_LEN=12.
// One wave32 owns 32 batch rows (two 16-row WMMA tiles) for the entire
// recurrence: gates(32x512) = [h|x|0](32x160) @ W^T via
// v_wmma_f32_16x16x32_bf16. Each LDS B-fragment read feeds TWO WMMAs
// (M-tile pair), doubling math per LDS byte vs a 16-row wave.
// 4 waves/block so work spreads across more WGPs (LDS bw is per-WGP).
// ---------------------------------------------------------------------------

typedef __attribute__((ext_vector_type(16))) __bf16        v16bf;
typedef __attribute__((ext_vector_type(8)))  __bf16        v8bf;
typedef __attribute__((ext_vector_type(8)))  float         v8f;
typedef __attribute__((ext_vector_type(4)))  unsigned int  u32x4;

#define LSTM_B        16384
#define LSTM_T        50
#define LSTM_I        5
#define LSTM_H        128
#define LSTM_PRED     12

#define NKB           5                 // K blocks of 32 (128 hidden + 32 aug)
#define WFRAGS        (NKB * 32)        // 160 B-fragments (kb x n-tile)
#define MT            2                 // M tiles per wave (32 batch rows)
#define STR_BF        176               // staging row stride (bf16 elems)
#define STR_B         (STR_BF * 2)      // 352 bytes, multiple of 16
#define WAVES_PER_BLK 4
#define BLK_THREADS   (WAVES_PER_BLK * 32)
#define ROWS_PER_WAVE (MT * 16)         // 32
#define STG_PER_WAVE  (ROWS_PER_WAVE * STR_B)   // 11264 B

#define OFF_W         ((size_t)0)                       // 160 KB fragments
#define OFF_BIAS      (OFF_W + (size_t)WFRAGS * 1024)   // f32[512]
#define OFF_LINW      (OFF_BIAS + 512 * 4)              // f32[256]
#define OFF_LINB      (OFF_LINW + 256 * 4)              // f32[2] (+pad)
#define OFF_STG       (OFF_LINB + 16)                   // 16B aligned
#define LDS_TOTAL     (OFF_STG + (size_t)WAVES_PER_BLK * STG_PER_WAVE) // ~212KB

__device__ __forceinline__ float fast_exp(float x) {
    return __builtin_amdgcn_exp2f(x * 1.44269504088896340736f);
}
__device__ __forceinline__ float fast_sigmoid(float x) {
    return __builtin_amdgcn_rcpf(1.0f + fast_exp(-x));
}
__device__ __forceinline__ float fast_tanh(float x) {
    return 1.0f - 2.0f * __builtin_amdgcn_rcpf(fast_exp(2.0f * x) + 1.0f);
}
__device__ __forceinline__ v8f splat8(float b) {
    v8f v = {b, b, b, b, b, b, b, b};
    return v;
}

// Stage [Whh | Wih | 0] (4H x 160) as WMMA bf16 B-fragments in LDS.
// Fragment f = kb*32 + nt is 1 KB: lane l holds 16 contiguous-K bf16 values
// for column N = nt*16 + (l&15), K = kb*32 + (l>=16 ? 16 : 0) .. +15
// (contiguous-K order, same as the A loads: lanes 0-15 K=0-15, lanes 16-31 K=16-31).
__device__ __forceinline__ void stage_weights(char* lds,
                                              const float* __restrict__ Whh,
                                              const float* __restrict__ Wih,
                                              const float* __restrict__ bih,
                                              const float* __restrict__ bhh,
                                              int tid) {
    const int wv = tid >> 5;
    const int ln = tid & 31;
    for (int f = wv; f < WFRAGS; f += WAVES_PER_BLK) {
        const int kb    = f >> 5;                 // 0..4
        const int nt    = f & 31;                 // 0..31
        const int n     = nt * 16 + (ln & 15);    // gate row 0..511
        const int kbase = kb * 32 + ((ln >> 4) << 4);
        __attribute__((aligned(16))) __bf16 tmp[16];
#pragma unroll
        for (int j = 0; j < 16; ++j) {
            const int k = kbase + j;
            float v;
            if (k < LSTM_H)               v = Whh[n * LSTM_H + k];
            else if (k < LSTM_H + LSTM_I) v = Wih[n * LSTM_I + (k - LSTM_H)];
            else                          v = 0.0f;
            tmp[j] = (__bf16)v;
        }
        u32x4* dst = (u32x4*)(lds + OFF_W + (size_t)f * 1024 + (size_t)ln * 32);
        dst[0] = *(const u32x4*)&tmp[0];
        dst[1] = *(const u32x4*)&tmp[8];
    }
    float* bias = (float*)(lds + OFF_BIAS);
    for (int n = tid; n < 4 * LSTM_H; n += BLK_THREADS) bias[n] = bih[n] + bhh[n];
}

// Load one 16x32 bf16 A-fragment from the row-major staging tile.
// A layout: lane&15 = M; lanes 0-15 hold K {0..15}, lanes 16-31
// hold K {16..31} -> two 16-byte LDS reads per lane.
__device__ __forceinline__ v16bf load_afrag(const char* stg, int lane, int kb) {
    const int m   = lane & 15;
    const int sel = lane >> 4;
    const char* base = stg + m * STR_B + kb * 64 + sel * 32;
    u32x4 lo = *(const u32x4*)(base);
    u32x4 hi = *(const u32x4*)(base + 16);
    v8bf l8 = __builtin_bit_cast(v8bf, lo);
    v8bf h8 = __builtin_bit_cast(v8bf, hi);
    return __builtin_shufflevector(l8, h8, 0, 1, 2, 3, 4, 5, 6, 7,
                                   8, 9, 10, 11, 12, 13, 14, 15);
}

// One LSTM step for the wave's 32-row batch tile (two 16-row M tiles).
// Each B-fragment (kb, gate-tile) is loaded ONCE and used for 2 WMMAs.
__device__ __forceinline__ void lstm_step(char* stg, const char* wlds,
                                          const float* bias, int lane,
                                          v8f (&cst)[MT][8]) {
    v16bf A[MT][NKB];
#pragma unroll
    for (int u = 0; u < MT; ++u)
#pragma unroll
        for (int kb = 0; kb < NKB; ++kb)
            A[u][kb] = load_afrag(stg + u * 16 * STR_B, lane, kb);

    const int sel = lane >> 4;
    const int nl  = lane & 15;
#pragma unroll
    for (int j = 0; j < 8; ++j) {
        v8f acc[MT][4];
#pragma unroll
        for (int g = 0; g < 4; ++g) {
            const float b = bias[(g * 8 + j) * 16 + nl];
#pragma unroll
            for (int u = 0; u < MT; ++u) acc[u][g] = splat8(b);
        }
#pragma unroll
        for (int g = 0; g < 4; ++g) {
#pragma unroll
            for (int kb = 0; kb < NKB; ++kb) {
                const int frag = kb * 32 + g * 8 + j;
                v16bf bf = *(const v16bf*)(wlds + (size_t)frag * 1024 +
                                           (size_t)lane * 32);
#pragma unroll
                for (int u = 0; u < MT; ++u)
                    acc[u][g] = __builtin_amdgcn_wmma_f32_16x16x32_bf16(
                        false, A[u][kb], false, bf, (short)0, acc[u][g],
                        false, false);
            }
        }
        // i/f/g/o gate tiles share the (lane, vgpr-row) element position with
        // c tile j of each M tile: pure per-lane elementwise, no shuffles.
#pragma unroll
        for (int u = 0; u < MT; ++u) {
#pragma unroll
            for (int r = 0; r < 8; ++r) {
                const float iv = fast_sigmoid(acc[u][0][r]);
                const float fv = fast_sigmoid(acc[u][1][r]);
                const float gv = fast_tanh(acc[u][2][r]);
                const float ov = fast_sigmoid(acc[u][3][r]);
                const float cn = fv * cst[u][j][r] + iv * gv;
                cst[u][j][r] = cn;
                const float hn = ov * fast_tanh(cn);
                // D layout: row M = r + 8*sel (within tile u), col = j*16 + nl
                *(__bf16*)(stg + (u * 16 + r + sel * 8) * STR_B +
                           (j * 16 + nl) * 2) = (__bf16)hn;
            }
        }
    }
}

__global__ void __launch_bounds__(BLK_THREADS, 1)
lstm_seq2seq_kernel(const float* __restrict__ x,
                    const float* __restrict__ eWih, const float* __restrict__ eWhh,
                    const float* __restrict__ ebih, const float* __restrict__ ebhh,
                    const float* __restrict__ dWih, const float* __restrict__ dWhh,
                    const float* __restrict__ dbih, const float* __restrict__ dbhh,
                    const float* __restrict__ linW, const float* __restrict__ linb,
                    float* __restrict__ out) {
    extern __shared__ char lds[];
    const int tid    = threadIdx.x;
    const int wv     = tid >> 5;
    const int lane   = tid & 31;
    const int batch0 = ((int)blockIdx.x * WAVES_PER_BLK + wv) * ROWS_PER_WAVE;

    // ---- stage encoder weights, bias, and the output projection ----
    stage_weights(lds, eWhh, eWih, ebih, ebhh, tid);
    {
        float* lw = (float*)(lds + OFF_LINW);
        for (int i = tid; i < 256; i += BLK_THREADS) lw[i] = linW[i];  // 2x128
        if (tid < 2) ((float*)(lds + OFF_LINB))[tid] = linb[tid];
    }
    __syncthreads();

    char* stg = lds + OFF_STG + (size_t)wv * STG_PER_WAVE;
    {   // zero h columns and augmented-zero columns (cols 133..159 stay 0)
        unsigned* p = (unsigned*)stg;
        for (int i = lane; i < (ROWS_PER_WAVE * STR_B) / 4; i += 32) p[i] = 0u;
    }
    v8f cst[MT][8];
#pragma unroll
    for (int u = 0; u < MT; ++u)
#pragma unroll
        for (int j = 0; j < 8; ++j) cst[u][j] = splat8(0.0f);

    const float* bias = (const float*)(lds + OFF_BIAS);
    const char*  wlds = lds + OFF_W;

    // ------------------------------ encoder ------------------------------
    for (int t = 0; t < LSTM_T; ++t) {
        // all 32 lanes: lane l stages x_t for batch row l of this wave
        {
            const float* xp =
                x + ((size_t)(batch0 + lane) * LSTM_T + t) * LSTM_I;
#pragma unroll
            for (int q = 0; q < LSTM_I; ++q)
                *(__bf16*)(stg + lane * STR_B + (LSTM_H + q) * 2) = (__bf16)xp[q];
            if (t + 1 < LSTM_T) __builtin_prefetch(xp + LSTM_I, 0, 1);
        }
        __builtin_amdgcn_wave_barrier();
        lstm_step(stg, wlds, bias, lane, cst);
        __builtin_amdgcn_wave_barrier();
    }
    // staging cols 128..132 now hold x[:,49,:] == dec_in0 (velocity+context);
    // cols 130..132 (static context) persist through the decoder.

    // ---- swap in decoder weights (same LDS region) ----
    __syncthreads();
    stage_weights(lds, dWhh, dWih, dbih, dbhh, tid);
    __syncthreads();

    // ------------------------------ decoder ------------------------------
    const float* lw = (const float*)(lds + OFF_LINW);
    const float* lb = (const float*)(lds + OFF_LINB);
    for (int t = 0; t < LSTM_PRED; ++t) {
        lstm_step(stg, wlds, bias, lane, cst);
        __builtin_amdgcn_wave_barrier();
        // lane l computes both outputs for its batch row l:
        // pred[l][jj] = lin_b[jj] + h[l,:] . lin_W[jj,:]
        const __bf16* hrow = (const __bf16*)(stg + lane * STR_B);
#pragma unroll
        for (int jj = 0; jj < 2; ++jj) {
            float acc = lb[jj];
            const float* wrow = lw + jj * LSTM_H;
#pragma unroll 8
            for (int k = 0; k < LSTM_H; ++k)
                acc += (float)hrow[k] * wrow[k];
            out[((size_t)(batch0 + lane) * LSTM_PRED + t) * 2 + jj] = acc;
            // feed back as next dec_in (cols 128/129)
            *(__bf16*)(stg + lane * STR_B + (LSTM_H + jj) * 2) = (__bf16)acc;
        }
        __builtin_amdgcn_wave_barrier();
    }
}

extern "C" void kernel_launch(void* const* d_in, const int* in_sizes, int n_in,
                              void* d_out, int out_size, void* d_ws, size_t ws_size,
                              hipStream_t stream) {
    (void)in_sizes; (void)n_in; (void)out_size; (void)d_ws; (void)ws_size;
    const float* x    = (const float*)d_in[0];
    const float* eWih = (const float*)d_in[1];
    const float* eWhh = (const float*)d_in[2];
    const float* ebih = (const float*)d_in[3];
    const float* ebhh = (const float*)d_in[4];
    const float* dWih = (const float*)d_in[5];
    const float* dWhh = (const float*)d_in[6];
    const float* dbih = (const float*)d_in[7];
    const float* dbhh = (const float*)d_in[8];
    const float* linW = (const float*)d_in[9];
    const float* linb = (const float*)d_in[10];
    float* out = (float*)d_out;

    (void)hipFuncSetAttribute((const void*)lstm_seq2seq_kernel,
                              hipFuncAttributeMaxDynamicSharedMemorySize,
                              (int)LDS_TOTAL);

    const int blocks = LSTM_B / (ROWS_PER_WAVE * WAVES_PER_BLK);  // 128
    lstm_seq2seq_kernel<<<dim3(blocks), dim3(BLK_THREADS), LDS_TOTAL, stream>>>(
        x, eWih, eWhh, ebih, ebhh, dWih, dWhh, dbih, dbhh, linW, linb, out);
}
